// BQuantConv1d_csr_10273561772171
// MI455X (gfx1250) — matrix operand order repaired and checked by reading, not compiled, not run on hardware
//
#include <hip/hip_runtime.h>

#define BITS 8
#define NF   4096
#define NX   4096
#define G    (NX / 8)      // 512
#define T    4096          // 2 * 2048

#define BM 128
#define BN 128
#define BK 64
#define KT (NX / BK)       // 64

typedef __attribute__((ext_vector_type(16))) __bf16 v16bf;
typedef __attribute__((ext_vector_type(8)))  float  v8f;

union Frag32 { v16bf v; uint4 q[2]; };

// -------- async global->LDS (CDNA5 GLOBAL_LOAD_ASYNC_TO_LDS_B128) ----------
#if defined(__has_builtin)
#  if __has_builtin(__builtin_amdgcn_global_load_async_to_lds_b128)
#    define USE_ASYNC 1
#  endif
#endif
#ifndef USE_ASYNC
#  define USE_ASYNC 0
#endif

#if USE_ASYNC
typedef int v4i __attribute__((vector_size(16)));
typedef __attribute__((address_space(1))) v4i gv4i;   // global (AS1)
typedef __attribute__((address_space(3))) v4i lv4i;   // LDS    (AS3)
__device__ __forceinline__ void async_b128(const void* g, void* l) {
    __builtin_amdgcn_global_load_async_to_lds_b128((gv4i*)g, (lv4i*)l, 0, 0);
}
// ASYNCcnt waits (immediate operand -> dedicated constant variants).
// Async loads complete in order; each wave issues exactly 8 async-b128 per
// tile, so "asynccnt <= 8" proves the older of two in-flight tiles landed.
__device__ __forceinline__ void wait_async0() {
#if __has_builtin(__builtin_amdgcn_s_wait_asynccnt)
    __builtin_amdgcn_s_wait_asynccnt(0);
#endif
}
__device__ __forceinline__ void wait_async8() {
#if __has_builtin(__builtin_amdgcn_s_wait_asynccnt)
    __builtin_amdgcn_s_wait_asynccnt(8);
#endif
}
#endif

__device__ __forceinline__ unsigned short f2bf(float f) {
    unsigned int u = __float_as_uint(f);
    u += 0x7FFFu + ((u >> 16) & 1u);      // round-to-nearest-even
    return (unsigned short)(u >> 16);
}

// ---------------------------------------------------------------------------
// Decode packed sign bytes + per-bit-plane scales into an effective bf16
// weight matrix W[f][n] = sum_k scale[k,f] * (+/-1).
// ---------------------------------------------------------------------------
__global__ void decode_weights(const int* __restrict__ binary,
                               const float* __restrict__ scale,
                               unsigned short* __restrict__ Wb) {
    int idx = blockIdx.x * blockDim.x + threadIdx.x;   // f*G + g
    if (idx >= NF * G) return;
    int f = idx >> 9;                                  // / G (G == 512)

    float w[8] = {0.f, 0.f, 0.f, 0.f, 0.f, 0.f, 0.f, 0.f};
#pragma unroll
    for (int k = 0; k < BITS; ++k) {
        int by = binary[k * (NF * G) + idx] & 0xFF;
        float sc = scale[k * NF + f];
#pragma unroll
        for (int j = 0; j < 8; ++j)
            w[j] += ((by >> (7 - j)) & 1) ? sc : -sc;
    }
    union { unsigned short h[8]; uint4 q; } pack;
#pragma unroll
    for (int j = 0; j < 8; ++j) pack.h[j] = f2bf(w[j]);
    *reinterpret_cast<uint4*>(Wb + (size_t)idx * 8) = pack.q;
}

// ---------------------------------------------------------------------------
// x: f32 -> bf16, 8 elements per thread.
// ---------------------------------------------------------------------------
__global__ void convert_x(const float* __restrict__ x,
                          unsigned short* __restrict__ xb, int n8) {
    int idx = blockIdx.x * blockDim.x + threadIdx.x;
    if (idx >= n8) return;
    const float4* xp = reinterpret_cast<const float4*>(x) + (size_t)idx * 2;
    float4 a = xp[0], b = xp[1];
    union { unsigned short h[8]; uint4 q; } pack;
    pack.h[0] = f2bf(a.x); pack.h[1] = f2bf(a.y);
    pack.h[2] = f2bf(a.z); pack.h[3] = f2bf(a.w);
    pack.h[4] = f2bf(b.x); pack.h[5] = f2bf(b.y);
    pack.h[6] = f2bf(b.z); pack.h[7] = f2bf(b.w);
    reinterpret_cast<uint4*>(xb)[idx] = pack.q;
}

// ---------------------------------------------------------------------------
// LDS-tiled bf16 WMMA GEMM: out[T x NF] = Xb[T x K] * Wb[NF x K]^T + bias.
// 128x128 block tile, BK=64, 8 wave32 waves; each wave -> 32x64 (2x4 WMMA
// tiles), v_wmma_f32_16x16x32_bf16. Triple-buffered async global->LDS
// pipeline: two tiles in flight on ASYNCcnt, partial waits (<=8).
// ---------------------------------------------------------------------------
__global__ __launch_bounds__(256)
void wmma_gemm_bf16(const unsigned short* __restrict__ Xb,
                    const unsigned short* __restrict__ Wb,
                    const float* __restrict__ bias,
                    float* __restrict__ out) {
#if USE_ASYNC
#define NBUF 3
#else
#define NBUF 2
#endif
    __shared__ __align__(16) unsigned short Abuf[NBUF][BM * BK];  // 16 KB each
    __shared__ __align__(16) unsigned short Bbuf[NBUF][BN * BK];  // 16 KB each

    const int tid  = threadIdx.x;
    const int lane = tid & 31;
    const int wave = tid >> 5;
    const int wm   = wave & 3;      // wave M offset: wm*32
    const int wn   = wave >> 2;     // wave N offset: wn*64
    const int lrow = lane & 15;
    const int hi   = lane >> 4;     // lane-group select (wave32 halves)

    const int m0 = blockIdx.y * BM;
    const int f0 = blockIdx.x * BN;

    const uint4* Xg = reinterpret_cast<const uint4*>(Xb);  // 512 uint4 per row
    const uint4* Wg = reinterpret_cast<const uint4*>(Wb);

    v8f acc[2][4];
#pragma unroll
    for (int mt = 0; mt < 2; ++mt)
#pragma unroll
        for (int nt = 0; nt < 4; ++nt)
            acc[mt][nt] = (v8f){0.f,0.f,0.f,0.f,0.f,0.f,0.f,0.f};

    auto computeTile = [&](int buf) {
        const uint4* A4 = reinterpret_cast<const uint4*>(&Abuf[buf][0]);
        const uint4* B4 = reinterpret_cast<const uint4*>(&Bbuf[buf][0]);
#pragma unroll
        for (int kk = 0; kk < 2; ++kk) {      // two K=32 WMMA steps per BK=64
            const int k0 = kk * 32;
            Frag32 afr[2], bfr[4];
#pragma unroll
            for (int mt = 0; mt < 2; ++mt) {
                // A 16x32 bf16: lane L holds row M=L%16; lanes 0-15 cover
                // K {0..7,16..23}, lanes 16-31 cover K {8..15,24..31}.
                int row  = wm * 32 + mt * 16 + lrow;
                int base = row * BK + k0 + hi * 8;
                afr[mt].q[0] = A4[base >> 3];
                afr[mt].q[1] = A4[(base + 16) >> 3];
            }
#pragma unroll
            for (int nt = 0; nt < 4; ++nt) {
                // B 32x16 bf16: lane = N column; lanes 0-15 K {0..7,16..23},
                // lanes 16-31 K {8..15,24..31} (same order as A).
                int frow = wn * 64 + nt * 16 + lrow;
                int base = frow * BK + k0 + hi * 8;
                bfr[nt].q[0] = B4[base >> 3];
                bfr[nt].q[1] = B4[(base + 16) >> 3];
            }
#pragma unroll
            for (int mt = 0; mt < 2; ++mt)
#pragma unroll
                for (int nt = 0; nt < 4; ++nt)
                    acc[mt][nt] = __builtin_amdgcn_wmma_f32_16x16x32_bf16(
                        false, afr[mt].v, false, bfr[nt].v,
                        (short)0, acc[mt][nt], false, false);
        }
    };

#if USE_ASYNC
    // ---- triple-buffered async global->LDS pipeline (2 tiles in flight) ----
    auto issueTile = [&](int kt, int buf) {
#pragma unroll
        for (int c = 0; c < 4; ++c) {
            int lin = c * 256 + tid;          // 1024 uint4 per 128x64 tile
            int row = lin >> 3, q = lin & 7;
            async_b128(Xg + (size_t)(m0 + row) * 512 + kt * 8 + q,
                       &Abuf[buf][(size_t)lin * 8]);
            async_b128(Wg + (size_t)(f0 + row) * 512 + kt * 8 + q,
                       &Bbuf[buf][(size_t)lin * 8]);
        }
    };

    issueTile(0, 0);
    issueTile(1, 1);
    wait_async8();       // tile 0 landed (tile 1 still in flight)
    __syncthreads();

    int buf = 0;
    for (int kt = 0; kt < KT; ++kt) {
        int nbuf = buf + 1; if (nbuf == NBUF) nbuf = 0;
        int ibuf = nbuf + 1; if (ibuf == NBUF) ibuf = 0;   // (buf+2)%3
        if (kt + 2 < KT) issueTile(kt + 2, ibuf);
        computeTile(buf);
        if (kt + 2 < KT)      wait_async8();   // tile kt+1 landed, kt+2 in flight
        else if (kt + 1 < KT) wait_async0();   // drain final in-flight tile
        __syncthreads();      // collective visibility; one barrier suffices
        buf = nbuf;
    }
#else
    // ---- fallback: double-buffered VGPR staging path ----
    uint4 ra[4], rb[4];
    auto loadTiles = [&](int kt) {
#pragma unroll
        for (int c = 0; c < 4; ++c) {
            int lin = c * 256 + tid;
            int row = lin >> 3, q = lin & 7;
            ra[c] = Xg[(size_t)(m0 + row) * 512 + kt * 8 + q];
            rb[c] = Wg[(size_t)(f0 + row) * 512 + kt * 8 + q];
        }
    };
    loadTiles(0);
    int buf = 0;
    for (int kt = 0; kt < KT; ++kt) {
        uint4* A4 = reinterpret_cast<uint4*>(&Abuf[buf][0]);
        uint4* B4 = reinterpret_cast<uint4*>(&Bbuf[buf][0]);
#pragma unroll
        for (int c = 0; c < 4; ++c) {
            A4[c * 256 + tid] = ra[c];
            B4[c * 256 + tid] = rb[c];
        }
        __syncthreads();
        if (kt + 1 < KT) loadTiles(kt + 1);
        computeTile(buf);
        __syncthreads();
        buf ^= 1;
    }
#endif

    // Epilogue: C/D layout -> VGPR v holds M = v + 8*(lane>=16), N = lane%16.
#pragma unroll
    for (int nt = 0; nt < 4; ++nt) {
        int f = f0 + wn * 64 + nt * 16 + lrow;
        float bv = bias[f];
#pragma unroll
        for (int mt = 0; mt < 2; ++mt) {
            int trow = m0 + wm * 32 + mt * 16 + hi * 8;
#pragma unroll
            for (int v = 0; v < 8; ++v)
                out[(size_t)(trow + v) * NF + f] = acc[mt][nt][v] + bv;
        }
    }
}

// ---------------------------------------------------------------------------
extern "C" void kernel_launch(void* const* d_in, const int* in_sizes, int n_in,
                              void* d_out, int out_size, void* d_ws, size_t ws_size,
                              hipStream_t stream) {
    const float* x      = (const float*)d_in[0];
    const int*   binary = (const int*)  d_in[1];
    const float* scale  = (const float*)d_in[2];
    const float* bias   = (const float*)d_in[3];
    float* out = (float*)d_out;

    unsigned short* xb = (unsigned short*)d_ws;          // T*NX bf16  (32 MB)
    unsigned short* wb = xb + (size_t)T * NX;            // NF*NX bf16 (32 MB)

    {
        int n = T * NX / 8;
        convert_x<<<(n + 255) / 256, 256, 0, stream>>>(x, xb, n);
    }
    {
        int n = NF * G;
        decode_weights<<<(n + 255) / 256, 256, 0, stream>>>(binary, scale, wb);
    }
    dim3 grid(NF / BN, T / BM);
    wmma_gemm_bf16<<<grid, 256, 0, stream>>>(xb, wb, bias, out);
}
